// ValueRNN_Timescales_18236431139462
// MI455X (gfx1250) — matrix operand order repaired and checked by reading, not compiled, not run on hardware
//
#include <hip/hip_runtime.h>

typedef __attribute__((ext_vector_type(16))) _Float16 v16h;
typedef __attribute__((ext_vector_type(8)))  _Float16 v8h;
typedef __attribute__((ext_vector_type(8)))  float    v8f;

#define T_STEPS 2048
#define BATCH   128
#define IN_DIM  32
#define HID     128
#define H16S    136   // padded f16 row stride (halves): 272B, 16B-aligned,
                      // bank = (4*row + c) mod 64 -> conflict-free b128 loads

// One persistent workgroup (8 wave32s) runs the whole sequential scan.
// Wave j owns hidden-column tile j (16 hidden units). Per step, each wave
// computes 8 output tiles (all batch rows) with v_wmma_f32_16x16x32_f16:
//   acc = X_t(16x32) * W_in^T + sum_k h16(16x32) * W_hh^T   (5 WMMAs/tile)
// then tanh + leaky blend, writes h back to LDS (f32 master + padded f16
// ping-pong), and does the H->1 value readout as 4 more WMMAs against
// value_W packed into column 0 of a B operand. X_{t+1} is prefetched into
// the WGP cache each step (global_prefetch) to keep X loads off the
// critical path.
__launch_bounds__(256)
__global__ void value_rnn_scan_kernel(const float* __restrict__ X,
                                      const float* __restrict__ W_in,
                                      const float* __restrict__ W_hh,
                                      const float* __restrict__ rnn_bias,
                                      const float* __restrict__ alpha,
                                      const float* __restrict__ value_W,
                                      const float* __restrict__ value_bias,
                                      float* __restrict__ out)
{
    // f16 copy of h, double-buffered, padded rows: 2*128*136*2 = 68 KB
    __shared__ _Float16 h16[2][BATCH * H16S];
    // f32 master copy of h; each (b,n) touched by exactly one lane: 64 KB
    __shared__ float h32[BATCH * HID];

    const int tid  = threadIdx.x;
    const int lane = tid & 31;
    const int wave = tid >> 5;        // column tile j (hidden block)
    const int half = lane >> 4;       // 0: lanes 0-15, 1: lanes 16-31
    const int l15  = lane & 15;

    const int n = wave * 16 + l15;    // hidden unit owned by this lane (B/C/D column)

    // ---- per-lane constants (live for the whole scan)
    const float alpha_n  = alpha[n];
    const float one_m_a  = 1.0f - alpha_n;
    const float bias_n   = rnn_bias[n];
    const float vbias    = value_bias[0];

    // ---- B operands, resident in VGPRs for all 2048 steps.
    // 16-bit B (KxN=32x16) layout: lanes 0-15 hold N=lane, K=0..7,16..23 in elems 0..15;
    // lanes 16-31 hold N=lane-16, K=8..15,24..31.
    v16h b_in;                         // W_in^T column n = W_in row n (K=32)
    {
        const float* row = W_in + n * IN_DIM;
        const int kb = half * 8;
        #pragma unroll
        for (int e = 0; e < 16; ++e) b_in[e] = (_Float16)row[kb + (e & 7) + (e >> 3) * 16];
    }
    v16h b_hh[4];                      // W_hh^T column n = W_hh row n (K=128)
    #pragma unroll
    for (int k = 0; k < 4; ++k) {
        const float* row = W_hh + n * HID;
        const int kb = k * 32 + half * 8;
        #pragma unroll
        for (int e = 0; e < 16; ++e) b_hh[k][e] = (_Float16)row[kb + (e & 7) + (e >> 3) * 16];
    }
    v16h b_vw[4];                      // value_W packed into B column N=0; K=128
    #pragma unroll
    for (int k = 0; k < 4; ++k) {
        const int kb = k * 32 + half * 8;
        #pragma unroll
        for (int e = 0; e < 16; ++e)
            b_vw[k][e] = (l15 == 0) ? (_Float16)value_W[kb + (e & 7) + (e >> 3) * 16] : (_Float16)0.0f;
    }

    // ---- h = 0
    for (int idx = tid; idx < BATCH * HID; idx += 256) h32[idx] = 0.0f;
    for (int idx = tid; idx < BATCH * H16S; idx += 256) h16[0][idx] = (_Float16)0.0f;
    __syncthreads();

    int cur = 0;
    for (int t = 0; t < T_STEPS; ++t) {
        const int nxt = cur ^ 1;
        const float* Xt = X + (long)t * (BATCH * IN_DIM);

        // Prefetch next step's X slice (16 KB = 256 threads * 64 B) into the
        // WGP cache; lowers to global_prefetch_b8, no counters, fully async.
        if (t + 1 < T_STEPS) {
            __builtin_prefetch(Xt + (BATCH * IN_DIM) + tid * 16, 0, 3);
        }

        // ---- recurrent GEMM + pointwise update, 8 row tiles per wave
        #pragma unroll 2
        for (int i = 0; i < 8; ++i) {
            const int bb  = i * 16 + l15;   // A-matrix row (batch index) for this lane
            const int off = half * 8;       // A layout: two 8-half runs per lane

            // A fragment of X_t (16x32): elems 0..7 -> K=off.., 8..15 -> K=off+16..
            v16h ax;
            const float* xr = Xt + bb * IN_DIM;
            #pragma unroll
            for (int e = 0; e < 8; ++e) {
                ax[e]     = (_Float16)xr[off + e];
                ax[e + 8] = (_Float16)xr[off + 16 + e];
            }
            v8f acc = {};
            acc = __builtin_amdgcn_wmma_f32_16x16x32_f16(false, ax, false, b_in,
                                                         (short)0, acc, false, false);
            #pragma unroll
            for (int k = 0; k < 4; ++k) {
                const _Float16* hrow = &h16[cur][bb * H16S + k * 32 + off];
                v8h lo = *(const v8h*)hrow;          // ds_load_b128, conflict-free
                v8h hi = *(const v8h*)(hrow + 16);   // ds_load_b128, conflict-free
                v16h ah = __builtin_shufflevector(lo, hi,
                         0,1,2,3,4,5,6,7,8,9,10,11,12,13,14,15);
                acc = __builtin_amdgcn_wmma_f32_16x16x32_f16(false, ah, false, b_hh[k],
                                                             (short)0, acc, false, false);
            }

            // tanh + leaky blend; C/D layout: lane half picks M=r or M=r+8
            #pragma unroll
            for (int r = 0; r < 8; ++r) {
                const int brow = i * 16 + r + half * 8;
                const float x  = acc[r] + bias_n;
                // tanh(x) = 1 - 2/(1 + exp2(x * 2*log2(e)))  (TRANS, co-exec with WMMA)
                const float e2 = __builtin_amdgcn_exp2f(x * 2.8853900817779268f);
                const float ht = 1.0f - 2.0f * __builtin_amdgcn_rcpf(e2 + 1.0f);
                const float hp = h32[brow * HID + n];
                const float hn = one_m_a * hp + alpha_n * ht;
                h32[brow * HID + n]       = hn;
                h16[nxt][brow * H16S + n] = (_Float16)hn;
            }
        }
        __syncthreads();   // h16[nxt] complete

        // ---- value readout V[t,b] = vbias + sum_n h[b,n]*vw[n] as 4 WMMAs.
        // Wave j reduces batch rows j*16..j*16+15; result lives in column N=0.
        {
            v8f vacc = {};
            const int bb  = wave * 16 + l15;
            const int off = half * 8;
            #pragma unroll
            for (int k = 0; k < 4; ++k) {
                const _Float16* hrow = &h16[nxt][bb * H16S + k * 32 + off];
                v8h lo = *(const v8h*)hrow;
                v8h hi = *(const v8h*)(hrow + 16);
                v16h ah = __builtin_shufflevector(lo, hi,
                         0,1,2,3,4,5,6,7,8,9,10,11,12,13,14,15);
                vacc = __builtin_amdgcn_wmma_f32_16x16x32_f16(false, ah, false, b_vw[k],
                                                              (short)0, vacc, false, false);
            }
            if (l15 == 0) {            // lanes 0 and 16 hold column N=0
                float* vout = out + (long)t * BATCH + wave * 16 + half * 8;
                #pragma unroll
                for (int r = 0; r < 8; ++r) vout[r] = vbias + vacc[r];
            }
        }
        // No extra barrier needed: next step writes h16[cur] (the other buffer)
        // and h32 is single-owner per lane.
        cur = nxt;
    }

    // ---- last_hidden (B,H) in f32, appended after V
    float* lh = out + (long)T_STEPS * BATCH;
    for (int idx = tid; idx < BATCH * HID; idx += 256) lh[idx] = h32[idx];
}

extern "C" void kernel_launch(void* const* d_in, const int* in_sizes, int n_in,
                              void* d_out, int out_size, void* d_ws, size_t ws_size,
                              hipStream_t stream) {
    (void)in_sizes; (void)n_in; (void)out_size; (void)d_ws; (void)ws_size;
    const float* X          = (const float*)d_in[0];
    const float* W_in       = (const float*)d_in[1];
    const float* W_hh       = (const float*)d_in[2];
    const float* rnn_bias   = (const float*)d_in[3];
    const float* alpha      = (const float*)d_in[4];
    const float* value_W    = (const float*)d_in[5];
    const float* value_bias = (const float*)d_in[6];
    float* out = (float*)d_out;

    value_rnn_scan_kernel<<<dim3(1), dim3(256), 0, stream>>>(
        X, W_in, W_hh, rnn_bias, alpha, value_W, value_bias, out);
}
